// PathSynthesis_75402445848860
// MI455X (gfx1250) — matrix operand order repaired and checked by reading, not compiled, not run on hardware
//
#include <hip/hip_runtime.h>
#include <hip/hip_bf16.h>

// ---------------------------------------------------------------------------
// Problem constants (from reference): H=8, B=1, K=16, L=256, D=32
// ---------------------------------------------------------------------------
#define HH 8
#define KK 16
#define LL 256
#define DD 32

typedef __attribute__((ext_vector_type(16))) __bf16 v16bf;
typedef __attribute__((ext_vector_type(8)))  float  v8f;

// ---------------- workspace layout (bytes) ----------------
constexpr size_t SZ_GXB  = (size_t)HH * KK * LL * DD * 2;   // gathered x2 features, bf16
constexpr size_t SZ_XAB  = (size_t)HH * KK * KK * DD * 2;   // anchor-anchor features, bf16
constexpr size_t SZ_WLN  = 96 * 32 * 2;                     // folded LN*proj weights, bf16 (padded K=96)
constexpr size_t SZ_KQ   = 32 * 64 * 2;                     // [k_w | q_w*scale], bf16
constexpr size_t SZ_QBUF = (size_t)HH * KK * LL * 4 * 4;    // per-head normalized q, f32
constexpr size_t SZ_TBUF = (size_t)HH * KK * LL * 3 * 4;    // per-head t, f32

constexpr size_t OFF_GXB  = 0;
constexpr size_t OFF_XAB  = OFF_GXB + SZ_GXB;
constexpr size_t OFF_WQ   = OFF_XAB + SZ_XAB;
constexpr size_t OFF_KQQ  = OFF_WQ  + SZ_WLN;
constexpr size_t OFF_WT   = OFF_KQQ + SZ_KQ;
constexpr size_t OFF_KQT  = OFF_WT  + SZ_WLN;
constexpr size_t OFF_BQ   = OFF_KQT + SZ_KQ;   // bias_q 32 f32
constexpr size_t OFF_CQ   = OFF_BQ  + 128;     // colsum_q 32 f32
constexpr size_t OFF_BT   = OFF_CQ  + 128;
constexpr size_t OFF_CT   = OFF_BT  + 128;
constexpr size_t OFF_QBUF = OFF_CT  + 128;
constexpr size_t OFF_TBUF = OFF_QBUF + SZ_QBUF;

// ---------------- small device helpers ----------------
struct Q4 { float w, x, y, z; };

__device__ __forceinline__ Q4 quat_mm_dev(Q4 a, Q4 b) {
  // reference: a_out = a.w*b.w - dot(a.v,b.v); bcd = cross(b.v,a.v) + a.w*b.v + b.w*a.v
  Q4 r;
  r.w = a.w * b.w - (a.x * b.x + a.y * b.y + a.z * b.z);
  r.x = (b.y * a.z - b.z * a.y) + a.w * b.x + b.w * a.x;
  r.y = (b.z * a.x - b.x * a.z) + a.w * b.y + b.w * a.y;
  r.z = (b.x * a.y - b.y * a.x) + a.w * b.z + b.w * a.z;
  return r;
}
__device__ __forceinline__ Q4 norm_quat_dev(Q4 q) {
  float n2 = q.w * q.w + q.x * q.x + q.y * q.y + q.z * q.z + 1e-6f;
  float inv = rsqrtf(n2);
  float s = (q.w >= 0.f) ? inv : -inv;
  Q4 r = { q.w * s, q.x * s, q.y * s, q.z * s };
  return r;
}
__device__ __forceinline__ void rot_dev(Q4 q, float tx, float ty, float tz,
                                        float& ox, float& oy, float& oz) {
  Q4 t4 = { 0.f, tx, ty, tz };
  Q4 qi = { q.w, -q.x, -q.y, -q.z };
  Q4 r = quat_mm_dev(quat_mm_dev(q, t4), qi);
  ox = r.x; oy = r.y; oz = r.z;
}
__device__ __forceinline__ float sigmoid_dev(float x) { return 1.f / (1.f + __expf(-x)); }

// per-wave LDS producer->consumer ordering (cross-lane, same wave)
__device__ __forceinline__ void wave_lds_sync() {
  asm volatile("s_wait_dscnt 0x0" ::: "memory");
  __builtin_amdgcn_wave_barrier();
}

// ---------------- WMMA fragment helpers (16x16x32 bf16, wave32) ----------------
// A (16Mx32K bf16): lane<16 -> M=lane, K=0..15;
//                   lane>=16 -> same M, K=16..31.
__device__ __forceinline__ v16bf load_a_frag(const __bf16* rows, int rowstride,
                                             int kt, int lane) {
  int m  = lane & 15;
  int kb = (lane & 16) ? 16 : 0;
  const __bf16* r = rows + m * rowstride + kt * 32;
  v16bf a;
#pragma unroll
  for (int e = 0; e < 16; ++e) {
    int k = kb + e;
    a[e] = r[k];
  }
  return a;
}
// B (32Kx16N bf16): lane<16 -> N=lane, K=0..15; lane>=16 -> N=lane-16, K=16..31.
// src is stored transposed: src[n][k] (i.e. rows of qx), rowstride 32.
__device__ __forceinline__ v16bf load_b_frag_lds(const __bf16* src, int lane) {
  int n  = lane & 15;
  int kb = (lane & 16) ? 16 : 0;
  const __bf16* r = src + n * 32 + kb;
  v16bf b;
#pragma unroll
  for (int e = 0; e < 16; ++e) b[e] = r[e];
  return b;
}
// B from plain row-major weights W[Kdim][N]
__device__ __forceinline__ v16bf load_b_frag_w(const __bf16* W, int N,
                                               int kt, int nt, int lane) {
  int n  = nt * 16 + (lane & 15);
  int kb = kt * 32 + ((lane & 16) ? 16 : 0);
  v16bf b;
#pragma unroll
  for (int e = 0; e < 16; ++e) b[e] = W[(size_t)(kb + e) * N + n];
  return b;
}
// D (16x16 f32): vgpr r -> M = r + 8*(lane>=16), N = lane&15
__device__ __forceinline__ void store_d_bf16(__bf16* tile, int noff, int lane, v8f acc) {
  int n  = noff + (lane & 15);
  int mb = (lane & 16) ? 8 : 0;
#pragma unroll
  for (int r = 0; r < 8; ++r) tile[(mb + r) * 32 + n] = (__bf16)acc[r];
}

__device__ __forceinline__ v8f wmma_bf16(v16bf a, v16bf b, v8f c) {
  return __builtin_amdgcn_wmma_f32_16x16x32_bf16(false, a, false, b, (short)0, c,
                                                 false, false);
}

// ---------------------------------------------------------------------------
// prep 1: fold LayerNorm into proj weights, pack [k_w | q_w*scale], all bf16
// ---------------------------------------------------------------------------
__global__ void prep_weights_kernel(
    const float* lnw_q, const float* lnb_q, const float* pw_q, const float* pb_q,
    const float* kw_q, const float* qw_q,
    const float* lnw_t, const float* lnb_t, const float* pw_t, const float* pb_t,
    const float* kw_t, const float* qw_t,
    __bf16* Wq, __bf16* KQq, float* biasq, float* csq,
    __bf16* Wt, __bf16* KQt, float* biast, float* cst) {
  int tid = threadIdx.x;
  for (int idx = tid; idx < 96 * 32; idx += 256) {
    int k = idx >> 5, o = idx & 31;
    float vq = (k < 68) ? lnw_q[k] * pw_q[k * 32 + o] : 0.f;
    float vt = (k < 67) ? lnw_t[k] * pw_t[k * 32 + o] : 0.f;
    Wq[idx] = (__bf16)vq;
    Wt[idx] = (__bf16)vt;
  }
  const float scale = 0.17677669529663687f;  // 1/sqrt(32)
  for (int idx = tid; idx < 32 * 64; idx += 256) {
    int k = idx >> 6, o = idx & 63;
    float vq = (o < 32) ? kw_q[k * 32 + o] : qw_q[k * 32 + (o - 32)] * scale;
    float vt = (o < 32) ? kw_t[k * 32 + o] : qw_t[k * 32 + (o - 32)] * scale;
    KQq[idx] = (__bf16)vq;
    KQt[idx] = (__bf16)vt;
  }
  if (tid < 32) {
    float cs = 0.f, bs = 0.f;
    for (int c = 0; c < 68; ++c) { cs += lnw_q[c] * pw_q[c * 32 + tid]; bs += lnb_q[c] * pw_q[c * 32 + tid]; }
    csq[tid] = cs; biasq[tid] = bs + pb_q[tid];
    cs = 0.f; bs = 0.f;
    for (int c = 0; c < 67; ++c) { cs += lnw_t[c] * pw_t[c * 32 + tid]; bs += lnb_t[c] * pw_t[c * 32 + tid]; }
    cst[tid] = cs; biast[tid] = bs + pb_t[tid];
  }
}

// ---------------------------------------------------------------------------
// prep 2: gather the 16 used rows of x2D, pack bf16.
// gxb[h][k][l][d] = x2D[h*32+d][A[k]][l] ; xab[h][i][j][d] = x2D[h*32+d][A[i]][A[j]]
// ---------------------------------------------------------------------------
__global__ void prep_gather_kernel(const float* x2D, const int* anch,
                                   __bf16* gxb, __bf16* xab) {
  int blk = blockIdx.x;
  int h = blk >> 4, k = blk & 15;
  __shared__ int As[16];
  if (threadIdx.x < 16) As[threadIdx.x] = anch[threadIdx.x];
  __syncthreads();
  int ak = As[k];
  int l = threadIdx.x;
#pragma unroll 4
  for (int d = 0; d < 32; ++d) {
    float v = x2D[(size_t)(h * 32 + d) * (LL * LL) + (size_t)ak * LL + l];
    gxb[((size_t)(h * 16 + k) * LL + l) * 32 + d] = (__bf16)v;
  }
  for (int idx = threadIdx.x; idx < 16 * 32; idx += 256) {
    int j = idx >> 5, d = idx & 31;
    float v = x2D[(size_t)(h * 32 + d) * (LL * LL) + (size_t)ak * LL + As[j]];
    xab[((size_t)(h * 16 + k) * 16 + j) * 32 + d] = (__bf16)v;
  }
}

// ---------------------------------------------------------------------------
// pass 1 (fused attention). One workgroup per (h,t); 8 waves; wave owns l-slice.
// Per l: 11x v_wmma_f32_16x16x32_bf16 (proj 6, k/q 4, gram 1) + softmax.
// ---------------------------------------------------------------------------
template <bool IS_T>
__global__ __launch_bounds__(256) void pass1_kernel(
    const float* __restrict__ quat,     // (H,K,L,4)
    const float* __restrict__ trans,    // (H,K,L,3) (t-pass)
    const float* __restrict__ qbuf_in,  // (H,K,L,4) normalized q (t-pass)
    const int*   __restrict__ anch,
    const __bf16* __restrict__ gxb, const __bf16* __restrict__ xab,
    const __bf16* __restrict__ Wln, const __bf16* __restrict__ KQ,
    const float* __restrict__ bias, const float* __restrict__ colsum,
    float* __restrict__ out_buf)        // q: (H,K,L,4); t: (H,K,L,3)
{
  constexpr int NW  = 8;
  constexpr int DIM = IS_T ? 67 : 68;

  __shared__ __bf16 xA_s[16 * 32];
  __shared__ float  quatA_s[16 * 4];
  __shared__ float  tA_s[16 * 4];
  __shared__ int    As[16];
  __shared__ __bf16 rows_s[NW][16 * 96];
  __shared__ __bf16 projt_s[NW][16 * 32];
  __shared__ __bf16 kxt_s[NW][16 * 32];
  __shared__ __bf16 qxt_s[NW][16 * 32];
  __shared__ float  step1_s[NW][16 * 4];
  __shared__ float  stats_s[NW][16 * 2];
  __shared__ float  atten_s[NW][16];

  const int h    = blockIdx.x >> 4;
  const int t    = blockIdx.x & 15;
  const int tid  = threadIdx.x;
  const int w    = tid >> 5;
  const int lane = tid & 31;
  const int hlf  = lane >> 4;   // 0 or 1
  const int mrow = lane & 15;

  if (tid < 16) As[tid] = anch[tid];
  __syncthreads();
  for (int idx = tid; idx < 16 * 32; idx += 256)
    xA_s[idx] = xab[(size_t)((h * 16 + t) * 16) * 32 + idx];
  if (tid < 16) {
    int a = As[tid];
    if (!IS_T) {
      const float* p = quat + ((size_t)(h * 16 + t) * LL + a) * 4;
      for (int c = 0; c < 4; ++c) quatA_s[tid * 4 + c] = p[c];
    } else {
      const float* p = qbuf_in + ((size_t)(h * 16 + t) * LL + a) * 4;
      for (int c = 0; c < 4; ++c) quatA_s[tid * 4 + c] = p[c];
      const float* pt = trans + ((size_t)(h * 16 + t) * LL + a) * 3;
      for (int c = 0; c < 3; ++c) tA_s[tid * 4 + c] = pt[c];
    }
  }
  __syncthreads();

  __bf16* rows  = rows_s[w];
  __bf16* projt = projt_s[w];
  __bf16* kxt   = kxt_s[w];
  __bf16* qxt   = qxt_s[w];
  float*  step1 = step1_s[w];
  float*  stats = stats_s[w];
  float*  atten = atten_s[w];

  // prefill constant part of feature rows: x1 (anchor-anchor) + zero pad 64..95
  {
#pragma unroll
    for (int e = 0; e < 16; ++e)
      rows[mrow * 96 + hlf * 16 + e] = xA_s[mrow * 32 + hlf * 16 + e];
#pragma unroll
    for (int e = 0; e < 16; ++e)
      rows[mrow * 96 + 64 + hlf * 16 + e] = (__bf16)0.f;
  }

  // preload constant weight B-fragments into registers
  v16bf wB[6];
#pragma unroll
  for (int kt = 0; kt < 3; ++kt)
#pragma unroll
    for (int nt = 0; nt < 2; ++nt)
      wB[kt * 2 + nt] = load_b_frag_w(Wln, 32, kt, nt, lane);
  v16bf kqB[4];
#pragma unroll
  for (int nt = 0; nt < 4; ++nt) kqB[nt] = load_b_frag_w(KQ, 64, 0, nt, lane);

  const float b0  = bias[mrow],        b1  = bias[16 + mrow];
  const float cs0 = colsum[mrow],      cs1 = colsum[16 + mrow];
  wave_lds_sync();

  for (int l = w; l < LL; l += NW) {
    // ---- per-l feature rows: x2 part ----
    {
      const __bf16* g = gxb + ((size_t)(h * 16 + mrow) * LL + l) * 32 + hlf * 16;
#pragma unroll
      for (int e = 0; e < 16; ++e) rows[mrow * 96 + 32 + hlf * 16 + e] = g[e];
    }
    // ---- step1 (quaternion / translation) ----
    if (hlf == 0) {
      if (!IS_T) {
        Q4 q1 = { quatA_s[mrow * 4 + 0], quatA_s[mrow * 4 + 1],
                  quatA_s[mrow * 4 + 2], quatA_s[mrow * 4 + 3] };
        const float* p2 = quat + ((size_t)(h * 16 + mrow) * LL + l) * 4;
        Q4 q2 = { p2[0], p2[1], p2[2], p2[3] };
        Q4 qs = norm_quat_dev(quat_mm_dev(q1, q2));
        step1[mrow * 4 + 0] = qs.w; step1[mrow * 4 + 1] = qs.x;
        step1[mrow * 4 + 2] = qs.y; step1[mrow * 4 + 3] = qs.z;
        rows[mrow * 96 + 64] = (__bf16)qs.w; rows[mrow * 96 + 65] = (__bf16)qs.x;
        rows[mrow * 96 + 66] = (__bf16)qs.y; rows[mrow * 96 + 67] = (__bf16)qs.z;
      } else {
        Q4 q1 = { quatA_s[mrow * 4 + 0], quatA_s[mrow * 4 + 1],
                  quatA_s[mrow * 4 + 2], quatA_s[mrow * 4 + 3] };
        const float* p2 = trans + ((size_t)(h * 16 + mrow) * LL + l) * 3;
        float rx, ry, rz;
        rot_dev(q1, p2[0], p2[1], p2[2], rx, ry, rz);
        float tsx = tA_s[mrow * 4 + 0] + rx;
        float tsy = tA_s[mrow * 4 + 1] + ry;
        float tsz = tA_s[mrow * 4 + 2] + rz;
        step1[mrow * 4 + 0] = tsx; step1[mrow * 4 + 1] = tsy; step1[mrow * 4 + 2] = tsz;
        rows[mrow * 96 + 64] = (__bf16)sigmoid_dev(tsx);
        rows[mrow * 96 + 65] = (__bf16)sigmoid_dev(tsy);
        rows[mrow * 96 + 66] = (__bf16)sigmoid_dev(tsz);
      }
    }
    wave_lds_sync();
    // ---- LayerNorm stats (mean, rstd) per row ----
    {
      float s = 0.f, ss = 0.f;
      int e0 = hlf * 34;
      int e1 = e0 + 34; if (e1 > DIM) e1 = DIM;
      for (int e = e0; e < e1; ++e) {
        float v = (float)rows[mrow * 96 + e];
        s += v; ss += v * v;
      }
      s  += __shfl_xor(s, 16);
      ss += __shfl_xor(ss, 16);
      if (hlf == 0) {
        float mean = s * (1.f / DIM);
        float var  = ss * (1.f / DIM) - mean * mean;
        stats[mrow * 2 + 0] = mean;
        stats[mrow * 2 + 1] = rsqrtf(var + 1e-5f);
      }
    }
    wave_lds_sync();
    // ---- proj: (16 x 96) @ (96 x 32), LN folded ----
    v8f acc0 = {}, acc1 = {};
#pragma unroll
    for (int kt = 0; kt < 3; ++kt) {
      v16bf a = load_a_frag(rows, 96, kt, lane);
      acc0 = wmma_bf16(a, wB[kt * 2 + 0], acc0);
      acc1 = wmma_bf16(a, wB[kt * 2 + 1], acc1);
    }
    {
      int mb = hlf * 8;
#pragma unroll
      for (int r = 0; r < 8; ++r) {
        float mean = stats[(mb + r) * 2 + 0];
        float rstd = stats[(mb + r) * 2 + 1];
        float p0 = acc0[r] * rstd - mean * rstd * cs0 + b0;
        float p1 = acc1[r] * rstd - mean * rstd * cs1 + b1;
        projt[(mb + r) * 32 + mrow]      = (__bf16)p0;
        projt[(mb + r) * 32 + 16 + mrow] = (__bf16)p1;
      }
    }
    wave_lds_sync();
    // ---- kx / qx: (16 x 32) @ (32 x 64) ----
    {
      v16bf ap = load_a_frag(projt, 32, 0, lane);
      v8f c0 = {}, c1 = {}, c2 = {}, c3 = {};
      c0 = wmma_bf16(ap, kqB[0], c0);
      c1 = wmma_bf16(ap, kqB[1], c1);
      c2 = wmma_bf16(ap, kqB[2], c2);
      c3 = wmma_bf16(ap, kqB[3], c3);
      store_d_bf16(kxt, 0, lane, c0);
      store_d_bf16(kxt, 16, lane, c1);
      store_d_bf16(qxt, 0, lane, c2);
      store_d_bf16(qxt, 16, lane, c3);
    }
    wave_lds_sync();
    // ---- gram: a[i][j] = kx_i . qx_j (scale folded into qx) ----
    v8f am = {};
    {
      v16bf ka = load_a_frag(kxt, 32, 0, lane);
      v16bf qb = load_b_frag_lds(qxt, lane);
      am = wmma_bf16(ka, qb, am);
    }
    // ---- softmax over rows i (per column j), then mean over columns j ----
    {
      float mx = -3.4e38f;
#pragma unroll
      for (int r = 0; r < 8; ++r) mx = fmaxf(mx, am[r]);
      mx = fmaxf(mx, __shfl_xor(mx, 16));
      float p[8], s = 0.f;
#pragma unroll
      for (int r = 0; r < 8; ++r) { p[r] = __expf(am[r] - mx); s += p[r]; }
      s += __shfl_xor(s, 16);
      float inv = 1.f / s;
      int mb = hlf * 8;
#pragma unroll
      for (int r = 0; r < 8; ++r) {
        float v = p[r] * inv;
        v += __shfl_xor(v, 1);
        v += __shfl_xor(v, 2);
        v += __shfl_xor(v, 4);
        v += __shfl_xor(v, 8);
        if (mrow == 0) atten[mb + r] = v * (1.f / 16.f);
      }
    }
    wave_lds_sync();
    // ---- weighted sum of step1 over i ----
    {
      float o0 = 0.f, o1 = 0.f, o2 = 0.f, o3 = 0.f;
#pragma unroll
      for (int i = 0; i < 16; ++i) {
        float wv = atten[i];
        o0 += wv * step1[i * 4 + 0];
        o1 += wv * step1[i * 4 + 1];
        o2 += wv * step1[i * 4 + 2];
        if (!IS_T) o3 += wv * step1[i * 4 + 3];
      }
      if (!IS_T) {
        Q4 qo = norm_quat_dev({ o0, o1, o2, o3 });
        if (lane == 0) {
          float* o = out_buf + ((size_t)(h * 16 + t) * LL + l) * 4;
          o[0] = qo.w; o[1] = qo.x; o[2] = qo.y; o[3] = qo.z;
        }
      } else {
        if (lane == 0) {
          float* o = out_buf + ((size_t)(h * 16 + t) * LL + l) * 3;
          o[0] = o0; o[1] = o1; o[2] = o2;
        }
      }
    }
    wave_lds_sync();
  }
}

// ---------------------------------------------------------------------------
// finalize q: ref = norm(mean_h q), mask anchors, q_est by recomputing step1
// ---------------------------------------------------------------------------
__global__ void q_finalize_kernel(const float* __restrict__ quat,
                                  const float* __restrict__ qbuf,
                                  const int* __restrict__ anch,
                                  float* __restrict__ q_out,
                                  float* __restrict__ q_est) {
  int k = blockIdx.x, l = threadIdx.x;
  __shared__ int As[16];
  if (threadIdx.x < 16) As[threadIdx.x] = anch[threadIdx.x];
  __syncthreads();
  float m0 = 0.f, m1 = 0.f, m2 = 0.f, m3 = 0.f;
  for (int h = 0; h < HH; ++h) {
    const float* p = qbuf + ((size_t)(h * 16 + k) * LL + l) * 4;
    m0 += p[0]; m1 += p[1]; m2 += p[2]; m3 += p[3];
  }
  Q4 ref = norm_quat_dev({ m0 * 0.125f, m1 * 0.125f, m2 * 0.125f, m3 * 0.125f });
  float est = 0.f;
  for (int h = 0; h < HH; ++h) {
    for (int i = 0; i < 16; ++i) {
      const float* p1 = quat + ((size_t)(h * 16 + k) * LL + As[i]) * 4;
      const float* p2 = quat + ((size_t)(h * 16 + i) * LL + l) * 4;
      Q4 qs = norm_quat_dev(quat_mm_dev({ p1[0], p1[1], p1[2], p1[3] },
                                        { p2[0], p2[1], p2[2], p2[3] }));
      float d0 = qs.w - ref.w, d1 = qs.x - ref.x, d2 = qs.y - ref.y, d3 = qs.z - ref.z;
      est += d0 * d0 + d1 * d1 + d2 * d2 + d3 * d3;
    }
  }
  est *= (1.f / 128.f);
  bool mask = (As[k] == l);
  float* o = q_out + ((size_t)k * LL + l) * 4;
  o[0] = mask ? 1.f : ref.w;
  o[1] = mask ? 0.f : ref.x;
  o[2] = mask ? 0.f : ref.y;
  o[3] = mask ? 0.f : ref.z;
  q_est[k * LL + l] = est;
}

// ---------------------------------------------------------------------------
// finalize t: ref = mean_h t, mask anchors, t_est by recomputing t_step1
// ---------------------------------------------------------------------------
__global__ void t_finalize_kernel(const float* __restrict__ trans,
                                  const float* __restrict__ qbuf,
                                  const float* __restrict__ tbuf,
                                  const int* __restrict__ anch,
                                  float* __restrict__ t_out,
                                  float* __restrict__ t_est) {
  int k = blockIdx.x, l = threadIdx.x;
  __shared__ int As[16];
  if (threadIdx.x < 16) As[threadIdx.x] = anch[threadIdx.x];
  __syncthreads();
  float m0 = 0.f, m1 = 0.f, m2 = 0.f;
  for (int h = 0; h < HH; ++h) {
    const float* p = tbuf + ((size_t)(h * 16 + k) * LL + l) * 3;
    m0 += p[0]; m1 += p[1]; m2 += p[2];
  }
  m0 *= 0.125f; m1 *= 0.125f; m2 *= 0.125f;
  float est = 0.f;
  for (int h = 0; h < HH; ++h) {
    for (int i = 0; i < 16; ++i) {
      const float* pq = qbuf + ((size_t)(h * 16 + k) * LL + As[i]) * 4;
      const float* p1 = trans + ((size_t)(h * 16 + k) * LL + As[i]) * 3;
      const float* p2 = trans + ((size_t)(h * 16 + i) * LL + l) * 3;
      float rx, ry, rz;
      rot_dev({ pq[0], pq[1], pq[2], pq[3] }, p2[0], p2[1], p2[2], rx, ry, rz);
      float d0 = (p1[0] + rx) - m0, d1 = (p1[1] + ry) - m1, d2 = (p1[2] + rz) - m2;
      est += d0 * d0 + d1 * d1 + d2 * d2;
    }
  }
  est *= (1.f / 128.f);
  bool mask = (As[k] == l);
  float* o = t_out + ((size_t)k * LL + l) * 3;
  o[0] = mask ? 0.f : m0;
  o[1] = mask ? 0.f : m1;
  o[2] = mask ? 0.f : m2;
  t_est[k * LL + l] = est;
}

// ---------------------------------------------------------------------------
extern "C" void kernel_launch(void* const* d_in, const int* in_sizes, int n_in,
                              void* d_out, int out_size, void* d_ws, size_t ws_size,
                              hipStream_t stream) {
  (void)in_sizes; (void)n_in; (void)out_size; (void)ws_size;
  const float* x2D   = (const float*)d_in[0];
  const float* quat  = (const float*)d_in[1];
  const float* trans = (const float*)d_in[2];
  const int*   anch  = (const int*)d_in[3];
  const float* lnw_q = (const float*)d_in[4];
  const float* lnb_q = (const float*)d_in[5];
  const float* pw_q  = (const float*)d_in[6];
  const float* pb_q  = (const float*)d_in[7];
  const float* kw_q  = (const float*)d_in[8];
  const float* qw_q  = (const float*)d_in[9];
  const float* lnw_t = (const float*)d_in[10];
  const float* lnb_t = (const float*)d_in[11];
  const float* pw_t  = (const float*)d_in[12];
  const float* pb_t  = (const float*)d_in[13];
  const float* kw_t  = (const float*)d_in[14];
  const float* qw_t  = (const float*)d_in[15];

  char* ws = (char*)d_ws;
  __bf16* gxb  = (__bf16*)(ws + OFF_GXB);
  __bf16* xab  = (__bf16*)(ws + OFF_XAB);
  __bf16* Wq   = (__bf16*)(ws + OFF_WQ);
  __bf16* KQq  = (__bf16*)(ws + OFF_KQQ);
  __bf16* Wt   = (__bf16*)(ws + OFF_WT);
  __bf16* KQt  = (__bf16*)(ws + OFF_KQT);
  float*  bq   = (float*)(ws + OFF_BQ);
  float*  cq   = (float*)(ws + OFF_CQ);
  float*  bt   = (float*)(ws + OFF_BT);
  float*  ct   = (float*)(ws + OFF_CT);
  float*  qbuf = (float*)(ws + OFF_QBUF);
  float*  tbuf = (float*)(ws + OFF_TBUF);

  float* out    = (float*)d_out;
  float* oq     = out;                    // (K,L,4)
  float* ot     = out + KK * LL * 4;      // (K,L,3)
  float* oqest  = out + KK * LL * 7;      // (K,L)
  float* otest  = out + KK * LL * 8;      // (K,L)

  prep_weights_kernel<<<1, 256, 0, stream>>>(
      lnw_q, lnb_q, pw_q, pb_q, kw_q, qw_q,
      lnw_t, lnb_t, pw_t, pb_t, kw_t, qw_t,
      Wq, KQq, bq, cq, Wt, KQt, bt, ct);
  prep_gather_kernel<<<HH * KK, 256, 0, stream>>>(x2D, anch, gxb, xab);

  pass1_kernel<false><<<HH * KK, 256, 0, stream>>>(
      quat, nullptr, nullptr, anch, gxb, xab, Wq, KQq, bq, cq, qbuf);
  q_finalize_kernel<<<KK, LL, 0, stream>>>(quat, qbuf, anch, oq, oqest);

  pass1_kernel<true><<<HH * KK, 256, 0, stream>>>(
      quat, trans, qbuf, anch, gxb, xab, Wt, KQt, bt, ct, tbuf);
  t_finalize_kernel<<<KK, LL, 0, stream>>>(trans, qbuf, tbuf, anch, ot, otest);
}
